// Encoder_36421322670332
// MI455X (gfx1250) — matrix operand order repaired and checked by reading, not compiled, not run on hardware
//
#include <hip/hip_runtime.h>

typedef __attribute__((ext_vector_type(16))) _Float16 v16h;
typedef __attribute__((ext_vector_type(8)))  _Float16 v8h;
typedef __attribute__((ext_vector_type(8)))  float    v8f;
typedef __attribute__((ext_vector_type(4)))  float    v4f;

#define TB    512   // timesteps
#define NB    1024  // batch
#define FIN   8     // input features
#define HID1  64    // layer-1 hidden
#define HID2  32    // layer-2 hidden
#define A1W   64    // staged A1 row width (h1 only; x comes from global)
#define A2W   96    // staged A2 row width ([h2(32) | h1(64)])

// split workgroup barrier: signal after LDS writes, wait before next-step reads
#define WG_BAR_SIGNAL() asm volatile("s_wait_dscnt 0\ns_barrier_signal -1" ::: "memory")
#define WG_BAR_WAIT()   asm volatile("s_barrier_wait -1" ::: "memory")

__device__ __forceinline__ float sigm(float v) {
  return __builtin_amdgcn_rcpf(1.0f + __expf(-v));
}
__device__ __forceinline__ float tanh_(float v) {
  float e = __expf(-2.0f * v);
  return (1.0f - e) * __builtin_amdgcn_rcpf(1.0f + e);
}

// A-matrix fragment from LDS: elems 0..15 = K(base..+15) contiguous.
__device__ __forceinline__ v16h fragA(const _Float16* p) {
  v8h lo = *(const v8h*)p;
  v8h hi = *(const v8h*)(p + 8);
  v16h r;
#pragma unroll
  for (int i = 0; i < 8; ++i) { r[i] = lo[i]; r[8 + i] = hi[i]; }
  return r;
}

// B-matrix fragment direct from global f32: 16 contiguous K values of one column.
__device__ __forceinline__ v16h fragB16(const float* p) {
  v4f a = *(const v4f*)(p);
  v4f b = *(const v4f*)(p + 4);
  v4f c = *(const v4f*)(p + 8);
  v4f d = *(const v4f*)(p + 12);
  v16h r;
#pragma unroll
  for (int i = 0; i < 4; ++i) {
    r[i]      = (_Float16)a[i];
    r[4 + i]  = (_Float16)b[i];
    r[8 + i]  = (_Float16)c[i];
    r[12 + i] = (_Float16)d[i];
  }
  return r;
}

extern "C" __global__ __launch_bounds__(192, 1)
void lstm2_encoder(const float* __restrict__ x,
                   const float* __restrict__ Wih1, const float* __restrict__ Whh1,
                   const float* __restrict__ bih1, const float* __restrict__ bhh1,
                   const float* __restrict__ Wih2, const float* __restrict__ Whh2,
                   const float* __restrict__ bih2, const float* __restrict__ bhh2,
                   float* __restrict__ out)
{
  // Ping-pong h staging. Every wave reads buf[s&1] and writes buf[s&1 ^ 1]:
  // no intra-step LDS dependencies, one barrier per step.
  __shared__ __align__(16) _Float16 A1buf[2][16 * A1W];
  __shared__ __align__(16) _Float16 A2buf[2][16 * A2W];

  const int tid  = threadIdx.x;
  const int wv   = tid >> 5;      // 0..3: layer-1 producers; 4..5: layer-2 consumers
  const int lane = tid & 31;
  const int l16  = lane & 15;
  const int lhi  = lane >> 4;     // A: K half; B: K half; C/D: row half
  const int mrow = 8 * lhi;       // this lane's C/D rows are mrow..mrow+7
  const int b0   = blockIdx.x * 16;

  for (int i = tid; i < 2 * 16 * A1W; i += 192) (&A1buf[0][0])[i] = (_Float16)0.0f;
  for (int i = tid; i < 2 * 16 * A2W; i += 192) (&A2buf[0][0])[i] = (_Float16)0.0f;

  v16h az; v8f zacc;
#pragma unroll
  for (int i = 0; i < 16; ++i) az[i] = (_Float16)0.0f;
#pragma unroll
  for (int i = 0; i < 8; ++i)  zacc[i] = 0.0f;

  __syncthreads();   // staging zeroed & visible

  if (wv < 4) {
    // =================== layer-1 producer wave (j-block = wv) ===================
    // Owns gate columns j = wv*16 + l16 with full i/f/g/o: N-tiles nt = g*4 + wv.
    v16h B1f[3][4];
#pragma unroll
    for (int kt = 0; kt < 2; ++kt)
#pragma unroll
      for (int g = 0; g < 4; ++g) {
        const int n = (g * 4 + wv) * 16 + l16;
        B1f[kt][g] = fragB16(Whh1 + (size_t)n * HID1 + kt * 32 + 16 * lhi);
      }
#pragma unroll
    for (int g = 0; g < 4; ++g) {
      // K-tile 2: K 64..71 = Wih1 row, rest 0. Lanes lhi=1 (K 80..95) all zero.
      const int n = (g * 4 + wv) * 16 + l16;
      v16h b = az;
      if (lhi == 0) {
        v4f a0 = *(const v4f*)(Wih1 + (size_t)n * FIN);
        v4f a1 = *(const v4f*)(Wih1 + (size_t)n * FIN + 4);
#pragma unroll
        for (int i = 0; i < 4; ++i) { b[i] = (_Float16)a0[i]; b[4 + i] = (_Float16)a1[i]; }
      }
      B1f[2][g] = b;
    }
    float bb1[4];
#pragma unroll
    for (int g = 0; g < 4; ++g) {
      const int n = g * 64 + wv * 16 + l16;
      bb1[g] = bih1[n] + bhh1[n];
    }
    float c1[8];
#pragma unroll
    for (int r = 0; r < 8; ++r) c1[r] = 0.0f;
    const int j1 = wv * 16 + l16;

    const float* xp = x + (size_t)(b0 + l16) * TB * FIN;
    // preload x(0) (lanes 0..15 carry row l16; x lands in A elems 0..7)
    v16h xa = az;
    if (lane < 16) {
      v4f x0 = *(const v4f*)(xp);
      v4f x1 = *(const v4f*)(xp + 4);
#pragma unroll
      for (int i = 0; i < 4; ++i) { xa[i] = (_Float16)x0[i]; xa[4 + i] = (_Float16)x1[i]; }
    }

    for (int s = 0; s <= TB; ++s) {
      const int p = s & 1;
      if (s < TB) {
        v8f acc[4];
#pragma unroll
        for (int q = 0; q < 4; ++q) acc[q] = zacc;
#pragma unroll
        for (int kt = 0; kt < 2; ++kt) {             // K tiles over h1(s-1)
          v16h a = fragA(&A1buf[p][l16 * A1W + kt * 32 + 16 * lhi]);
#pragma unroll
          for (int g = 0; g < 4; ++g)
            acc[g] = __builtin_amdgcn_wmma_f32_16x16x32_f16(
                false, a, false, B1f[kt][g], (short)0, acc[g], false, false);
        }
#pragma unroll
        for (int g = 0; g < 4; ++g)                  // K tile 2: x(s)
          acc[g] = __builtin_amdgcn_wmma_f32_16x16x32_f16(
              false, xa, false, B1f[2][g], (short)0, acc[g], false, false);

        // gate eval; h1(s) -> buf[p^1]
#pragma unroll
        for (int r = 0; r < 8; ++r) {
          float iv = sigm (acc[0][r] + bb1[0]);
          float fv = sigm (acc[1][r] + bb1[1]);
          float gv = tanh_(acc[2][r] + bb1[2]);
          float ov = sigm (acc[3][r] + bb1[3]);
          float c  = fv * c1[r] + iv * gv;
          c1[r] = c;
          float h  = ov * tanh_(c);
          const int m = mrow + r;
          _Float16 hh = (_Float16)h;
          A1buf[p ^ 1][m * A1W + j1]      = hh;
          A2buf[p ^ 1][m * A2W + 32 + j1] = hh;
        }
      }
      WG_BAR_SIGNAL();
      // overlap zone: fetch x(s+1) while waiting for the slowest wave
      if (s + 1 < TB) {
        v16h xn = az;
        if (lane < 16) {
          v4f x0 = *(const v4f*)(xp + (size_t)(s + 1) * FIN);
          v4f x1 = *(const v4f*)(xp + (size_t)(s + 1) * FIN + 4);
#pragma unroll
          for (int i = 0; i < 4; ++i) { xn[i] = (_Float16)x0[i]; xn[4 + i] = (_Float16)x1[i]; }
        }
        xa = xn;
      }
      WG_BAR_WAIT();
    }
  } else {
    // ============ layer-2 consumer wave (j2-block = wv-4), lags 1 step ============
    const int cw = wv - 4;                           // 0 or 1
    v16h B2f[3][4];
#pragma unroll
    for (int g = 0; g < 4; ++g) {
      const int n = (g * 2 + cw) * 16 + l16;
      B2f[0][g] = fragB16(Whh2 + (size_t)n * HID2 + 16 * lhi);
#pragma unroll
      for (int kt = 1; kt < 3; ++kt)
        B2f[kt][g] = fragB16(Wih2 + (size_t)n * HID1 + (kt - 1) * 32 + 16 * lhi);
    }
    float bb2[4];
#pragma unroll
    for (int g = 0; g < 4; ++g) {
      const int n2 = g * 32 + cw * 16 + l16;
      bb2[g] = bih2[n2] + bhh2[n2];
    }
    float c2[8], h2o[8];
#pragma unroll
    for (int r = 0; r < 8; ++r) { c2[r] = 0.0f; h2o[r] = 0.0f; }
    const int j2 = cw * 16 + l16;

    for (int s = 0; s <= TB; ++s) {
      const int p = s & 1;
      if (s >= 1) {                                  // computes time tau = s-1
        v8f acc[4];
#pragma unroll
        for (int q = 0; q < 4; ++q) acc[q] = zacc;
#pragma unroll
        for (int kt = 0; kt < 3; ++kt) {
          // kt0: h2(tau-1); kt1,2: h1(tau) -- all staged in buf[p]
          v16h a = fragA(&A2buf[p][l16 * A2W + kt * 32 + 16 * lhi]);
#pragma unroll
          for (int g = 0; g < 4; ++g)
            acc[g] = __builtin_amdgcn_wmma_f32_16x16x32_f16(
                false, a, false, B2f[kt][g], (short)0, acc[g], false, false);
        }
        // gate eval; h2(tau) -> buf[p^1] cols 0..31
#pragma unroll
        for (int r = 0; r < 8; ++r) {
          float iv = sigm (acc[0][r] + bb2[0]);
          float fv = sigm (acc[1][r] + bb2[1]);
          float gv = tanh_(acc[2][r] + bb2[2]);
          float ov = sigm (acc[3][r] + bb2[3]);
          float c  = fv * c2[r] + iv * gv;
          c2[r] = c;
          float h  = ov * tanh_(c);
          h2o[r] = h;
          A2buf[p ^ 1][(mrow + r) * A2W + j2] = (_Float16)h;
        }
      }
      WG_BAR_SIGNAL();
      WG_BAR_WAIT();
    }

    // final h2(TB-1) -> out[B, 1, EMB]
#pragma unroll
    for (int r = 0; r < 8; ++r)
      out[(size_t)(b0 + mrow + r) * HID2 + j2] = h2o[r];
  }
}

extern "C" void kernel_launch(void* const* d_in, const int* in_sizes, int n_in,
                              void* d_out, int out_size, void* d_ws, size_t ws_size,
                              hipStream_t stream) {
  (void)in_sizes; (void)n_in; (void)d_ws; (void)ws_size; (void)out_size;
  const float* x    = (const float*)d_in[0];
  const float* Wih1 = (const float*)d_in[1];
  const float* Whh1 = (const float*)d_in[2];
  const float* bih1 = (const float*)d_in[3];
  const float* bhh1 = (const float*)d_in[4];
  const float* Wih2 = (const float*)d_in[5];
  const float* Whh2 = (const float*)d_in[6];
  const float* bih2 = (const float*)d_in[7];
  const float* bhh2 = (const float*)d_in[8];
  float* out = (float*)d_out;

  lstm2_encoder<<<dim3(NB / 16), dim3(192), 0, stream>>>(
      x, Wih1, Whh1, bih1, bhh1, Wih2, Whh2, bih2, bhh2, out);
}
